// RF_vel_31928786878578
// MI455X (gfx1250) — compile-verified, matrix operand order checked by reading, not run on hardware
//
#include <hip/hip_runtime.h>
#include <math.h>

typedef __attribute__((ext_vector_type(16))) _Float16 v16h;
typedef __attribute__((ext_vector_type(8)))  float    v8f;

#define NF 64

// ---- fast math helpers (bare CDNA5 TRANS ops, no libm fixup sequences) ----
__device__ __forceinline__ float fast_rcp(float v) {
    return __builtin_amdgcn_rcpf(v);            // v_rcp_f32
}

__device__ __forceinline__ float fast_sqrt(float v) {
    return __builtin_amdgcn_sqrtf(v);           // v_sqrt_f32
}

__device__ __forceinline__ float fast_tanh(float v) {
#if __has_builtin(__builtin_amdgcn_tanhf)
    return __builtin_amdgcn_tanhf(v);           // v_tanh_f32
#else
    return 1.0f - 2.0f * fast_rcp(__expf(2.0f * v) + 1.0f);
#endif
}

__device__ __forceinline__ float silu_f(float v) {
#if __has_builtin(__builtin_amdgcn_tanhf)
    // x*sigmoid(x) = 0.5*x*(1 + tanh(x/2)) : one trans op
    return 0.5f * v * (1.0f + __builtin_amdgcn_tanhf(0.5f * v));
#else
    return v * fast_rcp(1.0f + __expf(-v));
#endif
}

// xor-lane shuffle via ds_swizzle immediate (group-of-32 mode)
template <int XOR>
__device__ __forceinline__ float swz(float v) {
    constexpr int imm = (XOR << 10) | 0x1F;     // and_mask=0x1f, or=0, xor=XOR
    return __int_as_float(__builtin_amdgcn_ds_swizzle(__float_as_int(v), imm));
}

// ---------------------------------------------------------------- utility
__global__ void k_zero(float* __restrict__ p, int n) {
    int i = blockIdx.x * blockDim.x + threadIdx.x;
    if (i < n) p[i] = 0.0f;
}

__global__ void k_count(const int* __restrict__ row, float* __restrict__ cnt, int M) {
    int e = blockIdx.x * blockDim.x + threadIdx.x;
    if (e < M) atomicAdd(&cnt[row[e]], 1.0f);
}

__global__ void k_invcnt(float* __restrict__ cnt, int n) {
    int i = blockIdx.x * blockDim.x + threadIdx.x;
    if (i < n) cnt[i] = fast_rcp(fmaxf(cnt[i], 1.0f));
}

// ---------------------------------------------------------------- edge MLP (WMMA)
// 16 edges per wave-iteration. A = [16 x 32] f16 (K0=radial,K1=ea0,K2=ea1,K3=1),
// B tiles = W1 rows + bias in K=0..3, N = 64 split into 4 tiles of 16.
__global__ void k_edge(const float* __restrict__ x,
                       const int* __restrict__ row, const int* __restrict__ col,
                       const float* __restrict__ eattr,
                       const float* __restrict__ w1,   // [3][64]
                       const float* __restrict__ b1,   // [64]
                       const float* __restrict__ w2,   // [64]
                       float* __restrict__ agg, int M) {
    const int lane = threadIdx.x & 31;
    const int half = lane >> 4;           // 0: row side, 1: col side
    const int sub  = lane & 15;
    const float lm = half ? 0.0f : 1.0f;  // only lanes 0-15 carry K=0..3
    // exactly one of {lane, lane^16} qualifies per edge: bit4 == bit3
    const bool do_atomic = (((lane >> 4) & 1) == ((lane >> 3) & 1));

    // Build the four 32x16 B tiles once (layer weights are kernel-constant).
    v16h B0 = {}, B1 = {}, B2 = {}, B3 = {};
    float w20, w21, w22, w23;
    {
        int n0 = sub, n1 = 16 + sub, n2 = 32 + sub, n3 = 48 + sub;
        B0[0]=(_Float16)(w1[n0]*lm);      B0[1]=(_Float16)(w1[NF+n0]*lm);
        B0[2]=(_Float16)(w1[2*NF+n0]*lm); B0[3]=(_Float16)(b1[n0]*lm);
        B1[0]=(_Float16)(w1[n1]*lm);      B1[1]=(_Float16)(w1[NF+n1]*lm);
        B1[2]=(_Float16)(w1[2*NF+n1]*lm); B1[3]=(_Float16)(b1[n1]*lm);
        B2[0]=(_Float16)(w1[n2]*lm);      B2[1]=(_Float16)(w1[NF+n2]*lm);
        B2[2]=(_Float16)(w1[2*NF+n2]*lm); B2[3]=(_Float16)(b1[n2]*lm);
        B3[0]=(_Float16)(w1[n3]*lm);      B3[1]=(_Float16)(w1[NF+n3]*lm);
        B3[2]=(_Float16)(w1[2*NF+n3]*lm); B3[3]=(_Float16)(b1[n3]*lm);
        w20 = w2[n0]; w21 = w2[n1]; w22 = w2[n2]; w23 = w2[n3];
    }

    const int wave   = (blockIdx.x * (blockDim.x >> 5)) + (threadIdx.x >> 5);
    const int nwaves = gridDim.x * (blockDim.x >> 5);
    const int ngrp   = (M + 15) >> 4;

    for (int g = wave; g < ngrp; g += nwaves) {
        int e = g * 16 + sub;
        float valid = (e < M) ? 1.0f : 0.0f;
        int ec = (e < M) ? e : 0;

        int r   = row[ec];
        int idx = half ? col[ec] : r;
        const float* p = x + (size_t)idx * 3;
        float vx = p[0], vy = p[1], vz = p[2];
        float eav = eattr[(size_t)ec * 2 + half];  // lower: ea0, upper: ea1

        float ox = swz<16>(vx);
        float oy = swz<16>(vy);
        float oz = swz<16>(vz);
        float oea = swz<16>(eav);                  // lower lanes receive ea1

        float dx = half ? (ox - vx) : (vx - ox);   // both halves: x[row]-x[col]
        float dy = half ? (oy - vy) : (vy - oy);
        float dz = half ? (oz - vz) : (vz - oz);
        float radial = fast_sqrt(dx*dx + dy*dy + dz*dz);

        v16h A = {};
        A[0] = (_Float16)(radial * lm);
        A[1] = (_Float16)(eav * lm);   // ea0 (lower half)
        A[2] = (_Float16)(oea * lm);   // ea1 (lower half)
        A[3] = (_Float16)lm;           // bias input = 1

        v8f c0 = {}, c1 = {}, c2 = {}, c3 = {};
        c0 = __builtin_amdgcn_wmma_f32_16x16x32_f16(false, A, false, B0, (short)0, c0, false, false);
        c1 = __builtin_amdgcn_wmma_f32_16x16x32_f16(false, A, false, B1, (short)0, c1, false, false);
        c2 = __builtin_amdgcn_wmma_f32_16x16x32_f16(false, A, false, B2, (short)0, c2, false, false);
        c3 = __builtin_amdgcn_wmma_f32_16x16x32_f16(false, A, false, B3, (short)0, c3, false, false);

        // stage 2: e_pre[m] = sum_n silu(h[m,n]) * w2[n]
        // lane holds n = sub+16t, m = j + 8*half (C layout: m-bit3 pinned to lane bit4)
#define PROW(j) (silu_f(c0[j])*w20 + silu_f(c1[j])*w21 + silu_f(c2[j])*w22 + silu_f(c3[j])*w23)
        float q0 = PROW(0), q1 = PROW(1), q2 = PROW(2), q3 = PROW(3);
        float q4 = PROW(4), q5 = PROW(5), q6 = PROW(6), q7 = PROW(7);
#undef PROW
        // Fused reduce-and-split: reduce over sub-lanes while assigning one m-bit
        // per lane bit; register count halves each step (8 ds ops total).
        // step 1: xor4, consume j2 (pairs q_k / q_{k+4}); lane bit2 <- m2
        bool b2 = (lane & 4) != 0;
        float r0 = (b2 ? q4 : q0) + swz<4>(b2 ? q0 : q4);
        float r1 = (b2 ? q5 : q1) + swz<4>(b2 ? q1 : q5);
        float r2 = (b2 ? q6 : q2) + swz<4>(b2 ? q2 : q6);
        float r3 = (b2 ? q7 : q3) + swz<4>(b2 ? q3 : q7);
        // step 2: xor2, consume j1; lane bit1 <- m1
        bool b1 = (lane & 2) != 0;
        float t0 = (b1 ? r2 : r0) + swz<2>(b1 ? r0 : r2);
        float t1 = (b1 ? r3 : r1) + swz<2>(b1 ? r1 : r3);
        // step 3: xor1, consume j0; lane bit0 <- m0
        bool b0 = (lane & 1) != 0;
        float u = (b0 ? t1 : t0) + swz<1>(b0 ? t0 : t1);
        // step 4: plain butterfly over xor8 (sum over s3; no m-bit left to split)
        u += swz<8>(u);
        // now lane holds e_pre[m = 8*bit4 + 4*bit2 + 2*bit1 + bit0];
        // lanes with bit4==bit3 satisfy m == sub (their own edge).

        float eout = fast_tanh(u) * valid;

        if (do_atomic) {
            float* a = agg + (size_t)r * 3;
            atomicAdd(a + 0, dx * eout);
            atomicAdd(a + 1, dy * eout);
            atomicAdd(a + 2, dz * eout);
        }
    }
}

// ---------------------------------------------------------------- per-node update
__global__ void k_update(const float* __restrict__ xin, float* __restrict__ agg,
                         const float* __restrict__ invc,
                         const float* __restrict__ vel, const float* __restrict__ veln,
                         const float* __restrict__ vw1, const float* __restrict__ vb1,
                         const float* __restrict__ vw2, const float* __restrict__ vb2,
                         float* __restrict__ xout, int N) {
    int i = blockIdx.x * blockDim.x + threadIdx.x;
    if (i >= N) return;
    float t = veln[i];
    float vs = vb2[0];
#pragma unroll 8
    for (int k = 0; k < NF; k++) vs += silu_f(t * vw1[k] + vb1[k]) * vw2[k];
    float ic = invc[i];
    size_t o = (size_t)i * 3;
    float ax = agg[o + 0] * ic, ay = agg[o + 1] * ic, az = agg[o + 2] * ic;
    agg[o + 0] = 0.0f; agg[o + 1] = 0.0f; agg[o + 2] = 0.0f;  // ready for next layer
    xout[o + 0] = xin[o + 0] + ax + vel[o + 0] * vs;
    xout[o + 1] = xin[o + 1] + ay + vel[o + 1] * vs;
    xout[o + 2] = xin[o + 2] + az + vel[o + 2] * vs;
}

// ---------------------------------------------------------------- launch
extern "C" void kernel_launch(void* const* d_in, const int* in_sizes, int n_in,
                              void* d_out, int out_size, void* d_ws, size_t ws_size,
                              hipStream_t stream) {
    const float* veln = (const float*)d_in[0];
    const float* x_in = (const float*)d_in[1];
    const int*   edges= (const int*)d_in[2];   // [2, M] int32
    const float* vel  = (const float*)d_in[3];
    const float* ea   = (const float*)d_in[4];
    const float* pw1  = (const float*)d_in[5]; // [L,3,64]
    const float* pb1  = (const float*)d_in[6]; // [L,64]
    const float* pw2  = (const float*)d_in[7]; // [L,64,1]
    const float* vw1  = (const float*)d_in[8]; // [L,1,64]
    const float* vb1  = (const float*)d_in[9]; // [L,64]
    const float* vw2  = (const float*)d_in[10];// [L,64,1]
    const float* vb2  = (const float*)d_in[11];// [L,1]

    const int M = in_sizes[2] / 2;
    const int N = in_sizes[0];
    const int* row = edges;
    const int* col = edges + M;

    float* ws   = (float*)d_ws;
    float* agg  = ws;                       // 3N
    float* cnt  = ws + (size_t)3 * N;       // N (becomes inv-count)
    float* xbuf = ws + (size_t)4 * N;       // 3N

    k_zero<<<(4 * N + 255) / 256, 256, 0, stream>>>(agg, 4 * N);
    k_count<<<(M + 255) / 256, 256, 0, stream>>>(row, cnt, M);
    k_invcnt<<<(N + 255) / 256, 256, 0, stream>>>(cnt, N);

    float* out = (float*)d_out;
    for (int l = 0; l < 4; l++) {
        const float* xs = (l == 0) ? x_in : xbuf;
        float*       xd = (l == 3) ? out : xbuf;
        k_edge<<<2048, 256, 0, stream>>>(xs, row, col, ea,
                                         pw1 + (size_t)l * 3 * NF, pb1 + (size_t)l * NF,
                                         pw2 + (size_t)l * NF, agg, M);
        k_update<<<(N + 255) / 256, 256, 0, stream>>>(xs, agg, cnt, vel, veln,
                                                      vw1 + (size_t)l * NF, vb1 + (size_t)l * NF,
                                                      vw2 + (size_t)l * NF, vb2 + l, xd, N);
    }
}
